// GShardMoE_73117523247424
// MI455X (gfx1250) — compile-verified, matrix operand order checked by reading, not run on hardware
//
#include <hip/hip_runtime.h>
#include <hip/hip_bf16.h>
#include <math.h>

#define D_    1024
#define E_    8
#define FF_   4096
#define CAP_  2048
#define T_    8192
#define NROWS (2 * E_ * CAP_)   // 32768 FFN rows (2 slots x 8 experts x 2048 cap)

typedef __attribute__((ext_vector_type(16))) __bf16 v16bf;
typedef __attribute__((ext_vector_type(8)))  __bf16 v8bf;
typedef __attribute__((ext_vector_type(8)))  float  v8f;
typedef int v4i_ __attribute__((vector_size(16)));   // type expected by async-LDS builtin

#if __has_builtin(__builtin_amdgcn_global_load_async_to_lds_b128) && \
    __has_builtin(__builtin_amdgcn_s_wait_asynccnt)
#define USE_ASYNC_LDS 1
#else
#define USE_ASYNC_LDS 0
#endif

#define GAS1(p) ((__attribute__((address_space(1))) v4i_*)(p))
#define LAS3(p) ((__attribute__((address_space(3))) v4i_*)(p))

__device__ __forceinline__ float gelu_tanh(float x) {
    float x3 = x * x * x;
    float inner = 0.7978845608028654f * (x + 0.044715f * x3);
    return 0.5f * x * (1.0f + tanhf(inner));
}

// ---------------------------------------------------------------------------
// 1) transpose + fp32->bf16 convert:  out[c][r] = (bf16) in[r][c], per expert
// ---------------------------------------------------------------------------
__global__ void transpose_cvt_kernel(const float* __restrict__ in,
                                     __bf16* __restrict__ out, int R, int C) {
    __shared__ float tile[32][33];
    const size_t eoff = (size_t)blockIdx.z * R * C;
    in  += eoff;
    out += eoff;
    const int r0 = blockIdx.y * 32, c0 = blockIdx.x * 32;
    for (int j = threadIdx.y; j < 32; j += 8)
        tile[j][threadIdx.x] = in[(size_t)(r0 + j) * C + c0 + threadIdx.x];
    __syncthreads();
    for (int j = threadIdx.y; j < 32; j += 8)
        out[(size_t)(c0 + j) * R + r0 + threadIdx.x] = (__bf16)tile[threadIdx.x][j];
}

// ---------------------------------------------------------------------------
// 2) gating: one wave (32 lanes) per token
// ---------------------------------------------------------------------------
__global__ __launch_bounds__(256) void gate_kernel(
    const float* __restrict__ x, const float* __restrict__ gw,
    float* __restrict__ p166o, float* __restrict__ probso, float* __restrict__ lse2o,
    int* __restrict__ e1o, int* __restrict__ e2o,
    float* __restrict__ g1no, float* __restrict__ g2no) {
    __shared__ float sgw[E_ * D_];                       // 32 KB
    for (int i = threadIdx.x; i < E_ * D_; i += 256) sgw[i] = gw[i];
    __syncthreads();

    const int lane = threadIdx.x & 31;
    const int wid  = threadIdx.x >> 5;
    const int t    = blockIdx.x * 8 + wid;
    const float* xr = x + (size_t)t * D_;

    float acc[E_];
#pragma unroll
    for (int e = 0; e < E_; ++e) acc[e] = 0.f;
    for (int d = lane; d < D_; d += 32) {
        float xv = xr[d];
#pragma unroll
        for (int e = 0; e < E_; ++e) acc[e] += xv * sgw[e * D_ + d];
    }
#pragma unroll
    for (int e = 0; e < E_; ++e)
        for (int off = 16; off; off >>= 1) acc[e] += __shfl_xor(acc[e], off, 32);

    if (lane == 0) {
        float m = acc[0];
#pragma unroll
        for (int e = 1; e < E_; ++e) m = fmaxf(m, acc[e]);
        float se = 0.f, se166 = 0.f;
        const float inv166 = 1.0f / 1.66f;
#pragma unroll
        for (int e = 0; e < E_; ++e) {
            se    += expf(acc[e] - m);
            se166 += expf((acc[e] - m) * inv166);
        }
        float lse    = m + logf(se);
        float lse166 = m * inv166 + logf(se166);
        lse2o[t] = lse * lse;

        float pr[E_];
#pragma unroll
        for (int e = 0; e < E_; ++e) {
            p166o[t * E_ + e] = expf(acc[e] * inv166 - lse166);
            pr[e] = expf(acc[e] - lse);
            probso[t * E_ + e] = pr[e];
        }
        int b1 = 0; float v1 = pr[0];
#pragma unroll
        for (int e = 1; e < E_; ++e) if (pr[e] > v1) { v1 = pr[e]; b1 = e; }
        int b2 = -1; float v2 = -1.f;
#pragma unroll
        for (int e = 0; e < E_; ++e)
            if (e != b1 && pr[e] > v2) { v2 = pr[e]; b2 = e; }
        float den = v1 + v2;
        e1o[t] = b1; e2o[t] = b2;
        g1no[t] = v1 / den; g2no[t] = v2 / den;
    }
}

// ---------------------------------------------------------------------------
// 3) deterministic tree reduction of per-token stats
// ---------------------------------------------------------------------------
__global__ __launch_bounds__(256) void reduce_kernel(
    const float* __restrict__ p166, const float* __restrict__ probs,
    const float* __restrict__ lse2, float* __restrict__ gstats) {
    __shared__ float s[256 * 18];
    float acc[18];
#pragma unroll
    for (int i = 0; i < 18; ++i) acc[i] = 0.f;
    for (int t = threadIdx.x; t < T_; t += 256) {
        float spa = 0.f;
#pragma unroll
        for (int e = 0; e < E_; ++e) {
            float p = p166[t * E_ + e];
            acc[e] += p;
            spa += p * (1.f - p);
            acc[8 + e] += probs[t * E_ + e];
        }
        acc[16] += spa;
        acc[17] += lse2[t];
    }
#pragma unroll
    for (int i = 0; i < 18; ++i) s[threadIdx.x * 18 + i] = acc[i];
    for (int off = 128; off > 0; off >>= 1) {
        __syncthreads();
        if (threadIdx.x < off)
            for (int i = 0; i < 18; ++i)
                s[threadIdx.x * 18 + i] += s[(threadIdx.x + off) * 18 + i];
    }
    if (threadIdx.x == 0)
        for (int i = 0; i < 18; ++i) gstats[i] = s[i];
}

// ---------------------------------------------------------------------------
// 4) single-wave ordered prefix scan for expert positions + aux finalize
// ---------------------------------------------------------------------------
__global__ void scan_finalize_kernel(
    const int* __restrict__ e1, const int* __restrict__ e2,
    const float* __restrict__ g1n, const float* __restrict__ g2n,
    const float* __restrict__ rnd, const float* __restrict__ gstats,
    int* __restrict__ row1, int* __restrict__ row2,
    float* __restrict__ wt1, float* __restrict__ wt2,
    int* __restrict__ use1, int* __restrict__ use2,
    float* __restrict__ auxOut) {
    const int lane = threadIdx.x;
    const unsigned ltm = (1u << lane) - 1u;
    int c1[E_], c2[E_];
    float impP[E_];
#pragma unroll
    for (int e = 0; e < E_; ++e) { c1[e] = 0; c2[e] = 0; impP[e] = 0.f; }

    for (int base = 0; base < T_; base += 32) {
        int t = base + lane;
        int a = e1[t], b = e2[t];
        float ga = g1n[t], gb = g2n[t];
        int p1 = 0, p2 = 0;
#pragma unroll
        for (int e = 0; e < E_; ++e) {
            unsigned m1 = (unsigned)__ballot(a == e);
            unsigned m2 = (unsigned)__ballot(b == e);
            if (a == e) p1 = c1[e] + __popc(m1 & ltm);
            if (b == e) p2 = c2[e] + __popc(m2 & ltm);
            c1[e] += __popc(m1);
            c2[e] += __popc(m2);
        }
        bool ok1 = p1 < CAP_;
        bool ok2 = (p2 < CAP_) && (rnd[t] < 2.f * gb);
        float w1v = ok1 ? ga : 0.f;
        float w2v = ok2 ? gb : 0.f;
        int pc1 = p1 < CAP_ - 1 ? p1 : CAP_ - 1;
        int pc2 = p2 < CAP_ - 1 ? p2 : CAP_ - 1;
        row1[t] = a * CAP_ + pc1;
        row2[t] = E_ * CAP_ + b * CAP_ + pc2;
        use1[t] = ok1 ? 1 : 0;
        use2[t] = ok2 ? 1 : 0;
        wt1[t] = w1v;
        wt2[t] = w2v;
#pragma unroll
        for (int e = 0; e < E_; ++e) {
            impP[e] += (a == e) ? w1v : 0.f;
            impP[e] += (b == e) ? w2v : 0.f;
        }
    }
    float imp[E_];
#pragma unroll
    for (int e = 0; e < E_; ++e) {
        float v = impP[e];
        for (int off = 16; off; off >>= 1) v += __shfl_xor(v, off, 32);
        imp[e] = v;
    }
    if (lane == 0) {
        const float Tf = (float)T_;
        float pa = gstats[16] / (Tf * 8.f);
        float pb = 0.f;
#pragma unroll
        for (int e = 0; e < E_; ++e) { float pm = gstats[e] / Tf; pb += pm * (1.f - pm); }
        pb = 1.f / 8.f - pb / 8.f;
        float penalty = 0.01f * (pa + pb);
        float zl = 0.001f * (gstats[17] / Tf);
        float load = 0.f;
#pragma unroll
        for (int e = 0; e < E_; ++e) load += ((float)c1[e] / Tf) * (gstats[8 + e] / Tf);
        load *= 0.01f * 8.f;
        float mean = 0.f;
#pragma unroll
        for (int e = 0; e < E_; ++e) mean += imp[e];
        mean *= (1.f / 8.f);
        float var = 0.f;
#pragma unroll
        for (int e = 0; e < E_; ++e) { float d = imp[e] - mean; var += d * d; }
        var *= (1.f / 8.f);
        float il = 0.01f * var / (mean * mean);
        auxOut[0] = penalty + zl + load + il;
    }
}

// ---------------------------------------------------------------------------
// 5) zero + scatter dispatch (fp32 -> bf16)
// ---------------------------------------------------------------------------
__global__ void zero_kernel(uint4* __restrict__ p, size_t n) {
    size_t i = (size_t)blockIdx.x * 256 + threadIdx.x;
    if (i < n) p[i] = uint4{0u, 0u, 0u, 0u};
}

__global__ __launch_bounds__(128) void dispatch_kernel(
    const float* __restrict__ x, const int* __restrict__ row1, const int* __restrict__ row2,
    const int* __restrict__ use1, const int* __restrict__ use2, __bf16* __restrict__ Buf) {
    const int t = blockIdx.x, slot = blockIdx.y;
    const int use = slot ? use2[t] : use1[t];
    if (!use) return;
    const int row = slot ? row2[t] : row1[t];
    const float4* xr = (const float4*)(x + (size_t)t * D_);
    float4 a = xr[threadIdx.x * 2];
    float4 b = xr[threadIdx.x * 2 + 1];
    v8bf o = { (__bf16)a.x, (__bf16)a.y, (__bf16)a.z, (__bf16)a.w,
               (__bf16)b.x, (__bf16)b.y, (__bf16)b.z, (__bf16)b.w };
    *(v8bf*)(Buf + (size_t)row * D_ + threadIdx.x * 8) = o;
}

// ---------------------------------------------------------------------------
// 6) bf16 WMMA GEMM: C[M,N] = A[M,K] * W[e][N,K]^T + bias[e][N]
//    block tile 256x128, 8 waves (4Mx2N) of 64x64 each -> 16 WMMA / 16 ds_load
//    per K=32 step. Async double-buffered LDS; last K-step peeled so the main
//    loop is branch-free. MODE 0: gelu -> bf16 (H); MODE 1: f32 (Out)
// ---------------------------------------------------------------------------
#define LSTR  48            // padded LDS row stride (elems): 96B rows, 16B groups
#define ASZ   (256 * LSTR)  // A tile elems per buffer
#define BSZ   (128 * LSTR)  // B tile elems per buffer

template <int MODE>
__global__ __launch_bounds__(256) void gemm_kernel(
    const __bf16* __restrict__ A, const __bf16* __restrict__ W,
    const float* __restrict__ bias, void* __restrict__ outP, int N, int K) {
#if USE_ASYNC_LDS
    __shared__ __align__(16) __bf16 sA[2 * ASZ];
    __shared__ __align__(16) __bf16 sB[2 * BSZ];
#else
    __shared__ __align__(16) __bf16 sA[ASZ];
    __shared__ __align__(16) __bf16 sB[BSZ];
#endif

    const int m0 = blockIdx.y * 256;
    const int n0 = blockIdx.x * 128;
    const int e  = (m0 >> 11) & 7;                 // 2048 rows per expert block
    const __bf16* Wp = W + (size_t)e * N * K;
    const float*  bp = bias + e * N;

    const int tid  = threadIdx.x;
    const int lane = tid & 31;
    const int wid  = tid >> 5;
    const int wm   = wid & 3;        // 4 waves along M (64 rows each)
    const int wn   = wid >> 2;       // 2 waves along N (64 cols each)
    const int lh   = lane >> 4;      // selects K groups {0..7,16..23} vs {8..15,24..31}
    const int lm   = lane & 15;

    const int rA = tid >> 2, gA = tid & 3;         // base (row,group) for loads

    v8f acc[4][4] = {};

    // compute one K=32 step from LDS buffers bA/bB
    auto compute = [&](const __bf16* bA, const __bf16* bB) {
        v16bf afrag[4], bfrag[4];
#pragma unroll
        for (int mi = 0; mi < 4; ++mi) {
            const __bf16* pa = &bA[(wm * 64 + mi * 16 + lm) * LSTR + lh * 8];
            v8bf lo = *(const v8bf*)pa;
            v8bf hi = *(const v8bf*)(pa + 16);
            afrag[mi] = __builtin_shufflevector(lo, hi,
                0,1,2,3,4,5,6,7,8,9,10,11,12,13,14,15);
        }
#pragma unroll
        for (int ni = 0; ni < 4; ++ni) {
            const __bf16* pb = &bB[(wn * 64 + ni * 16 + lm) * LSTR + lh * 8];
            v8bf lo = *(const v8bf*)pb;
            v8bf hi = *(const v8bf*)(pb + 16);
            bfrag[ni] = __builtin_shufflevector(lo, hi,
                0,1,2,3,4,5,6,7,8,9,10,11,12,13,14,15);
        }
#pragma unroll
        for (int mi = 0; mi < 4; ++mi)
#pragma unroll
            for (int ni = 0; ni < 4; ++ni)
                acc[mi][ni] = __builtin_amdgcn_wmma_f32_16x16x32_bf16(
                    false, afrag[mi], false, bfrag[ni],
                    (short)0, acc[mi][ni], false, false);
    };

#if USE_ASYNC_LDS
    auto issue = [&](int k, int buf) {
#pragma unroll
        for (int i = 0; i < 4; ++i) {              // A: 256 rows x 4 groups
            int row = rA + i * 64;
            __builtin_amdgcn_global_load_async_to_lds_b128(
                GAS1(A + (size_t)(m0 + row) * K + k + gA * 8),
                LAS3(sA + buf * ASZ + row * LSTR + gA * 8),
                0, 0);
        }
#pragma unroll
        for (int i = 0; i < 2; ++i) {              // B: 128 rows x 4 groups
            int row = rA + i * 64;
            __builtin_amdgcn_global_load_async_to_lds_b128(
                GAS1(Wp + (size_t)(n0 + row) * K + k + gA * 8),
                LAS3(sB + buf * BSZ + row * LSTR + gA * 8),
                0, 0);
        }
    };
    issue(0, 0);
    int cur = 0;
    // steady state: branch-free, always prefetch next tile into the other buffer
    for (int k = 0; k + 32 < K; k += 32) {
        issue(k + 32, cur ^ 1);
        __builtin_amdgcn_s_wait_asynccnt(6);       // in-order: tile k complete
        __syncthreads();
        compute(sA + cur * ASZ, sB + cur * BSZ);
        __syncthreads();                           // buf 'cur' free for re-issue
        cur ^= 1;
    }
    // peeled last tile
    __builtin_amdgcn_s_wait_asynccnt(0);
    __syncthreads();
    compute(sA + cur * ASZ, sB + cur * BSZ);
#else
    for (int k = 0; k < K; k += 32) {
        uint4 ra[4], rb[2];
#pragma unroll
        for (int i = 0; i < 4; ++i)
            ra[i] = *(const uint4*)&A[(size_t)(m0 + rA + i * 64) * K + k + gA * 8];
#pragma unroll
        for (int i = 0; i < 2; ++i)
            rb[i] = *(const uint4*)&Wp[(size_t)(n0 + rA + i * 64) * K + k + gA * 8];
        __syncthreads();
#pragma unroll
        for (int i = 0; i < 4; ++i)
            *(uint4*)&sA[(rA + i * 64) * LSTR + gA * 8] = ra[i];
#pragma unroll
        for (int i = 0; i < 2; ++i)
            *(uint4*)&sB[(rA + i * 64) * LSTR + gA * 8] = rb[i];
        __syncthreads();
        compute(sA, sB);
    }
#endif

#pragma unroll
    for (int mi = 0; mi < 4; ++mi) {
#pragma unroll
        for (int ni = 0; ni < 4; ++ni) {
            const int gcol = n0 + wn * 64 + ni * 16 + lm;
            const float bv = bp[gcol];
#pragma unroll
            for (int i = 0; i < 8; ++i) {
                const int grow = m0 + wm * 64 + mi * 16 + lh * 8 + i;
                const size_t gi = (size_t)grow * N + gcol;
                float v = acc[mi][ni][i] + bv;
                if (MODE == 0) ((__bf16*)outP)[gi] = (__bf16)gelu_tanh(v);
                else           ((float*)outP)[gi]  = v;
            }
        }
    }
}

// ---------------------------------------------------------------------------
// 7) combine: y[t] = Out[row1[t]] * wt1[t] + Out[row2[t]] * wt2[t]
// ---------------------------------------------------------------------------
__global__ __launch_bounds__(256) void combine_kernel(
    const float* __restrict__ Out, const int* __restrict__ row1, const int* __restrict__ row2,
    const float* __restrict__ wt1, const float* __restrict__ wt2, float* __restrict__ y) {
    const int t = blockIdx.x;
    const float a = wt1[t], b = wt2[t];
    const float4* r1 = (const float4*)(Out + (size_t)row1[t] * D_);
    const float4* r2 = (const float4*)(Out + (size_t)row2[t] * D_);
    float4* yr = (float4*)(y + (size_t)t * D_);
    float4 u = r1[threadIdx.x];
    float4 v = r2[threadIdx.x];
    float4 o;
    o.x = u.x * a + v.x * b;
    o.y = u.y * a + v.y * b;
    o.z = u.z * a + v.z * b;
    o.w = u.w * a + v.w * b;
    yr[threadIdx.x] = o;
}

// ---------------------------------------------------------------------------
// host launcher
// ---------------------------------------------------------------------------
extern "C" void kernel_launch(void* const* d_in, const int* in_sizes, int n_in,
                              void* d_out, int out_size, void* d_ws, size_t ws_size,
                              hipStream_t stream) {
    const float* x   = (const float*)d_in[0];
    const float* gw  = (const float*)d_in[1];
    const float* w1  = (const float*)d_in[2];
    const float* b1  = (const float*)d_in[3];
    const float* w2  = (const float*)d_in[4];
    const float* b2  = (const float*)d_in[5];
    const float* rnd = (const float*)d_in[6];

    char* p = (char*)d_ws;
    auto take = [&p](size_t bytes) -> void* {
        void* r = (void*)p;
        p += (bytes + 255) & ~(size_t)255;
        return r;
    };
    __bf16* Buf  = (__bf16*)take((size_t)NROWS * D_ * 2);
    __bf16* w1T  = (__bf16*)take((size_t)E_ * FF_ * D_ * 2);
    __bf16* w2T  = (__bf16*)take((size_t)E_ * D_ * FF_ * 2);
    __bf16* H    = (__bf16*)take((size_t)NROWS * FF_ * 2);
    float*  Outb = (float*) take((size_t)NROWS * D_ * 4);
    float*  p166 = (float*) take((size_t)T_ * E_ * 4);
    float*  prob = (float*) take((size_t)T_ * E_ * 4);
    float*  lse2 = (float*) take((size_t)T_ * 4);
    int*    e1   = (int*)   take((size_t)T_ * 4);
    int*    e2   = (int*)   take((size_t)T_ * 4);
    float*  g1n  = (float*) take((size_t)T_ * 4);
    float*  g2n  = (float*) take((size_t)T_ * 4);
    int*    row1 = (int*)   take((size_t)T_ * 4);
    int*    row2 = (int*)   take((size_t)T_ * 4);
    float*  wt1  = (float*) take((size_t)T_ * 4);
    float*  wt2  = (float*) take((size_t)T_ * 4);
    int*    use1 = (int*)   take((size_t)T_ * 4);
    int*    use2 = (int*)   take((size_t)T_ * 4);
    float*  gstats = (float*)take(18 * 4);

    float* y   = (float*)d_out;
    float* aux = y + (size_t)T_ * D_;

    transpose_cvt_kernel<<<dim3(FF_ / 32, D_ / 32, E_), dim3(32, 8), 0, stream>>>(w1, w1T, D_, FF_);
    transpose_cvt_kernel<<<dim3(D_ / 32, FF_ / 32, E_), dim3(32, 8), 0, stream>>>(w2, w2T, FF_, D_);

    const size_t nvec = (size_t)NROWS * D_ * 2 / 16;
    zero_kernel<<<dim3((unsigned)(nvec / 256)), dim3(256), 0, stream>>>((uint4*)Buf, nvec);

    gate_kernel<<<dim3(T_ / 8), dim3(256), 0, stream>>>(x, gw, p166, prob, lse2, e1, e2, g1n, g2n);
    reduce_kernel<<<dim3(1), dim3(256), 0, stream>>>(p166, prob, lse2, gstats);
    scan_finalize_kernel<<<dim3(1), dim3(32), 0, stream>>>(e1, e2, g1n, g2n, rnd, gstats,
                                                           row1, row2, wt1, wt2, use1, use2, aux);

    dispatch_kernel<<<dim3(T_, 2), dim3(128), 0, stream>>>(x, row1, row2, use1, use2, Buf);

    gemm_kernel<0><<<dim3(FF_ / 128, NROWS / 256), dim3(256), 0, stream>>>(
        Buf, w1T, b1, (void*)H, FF_, D_);
    gemm_kernel<1><<<dim3(D_ / 128, NROWS / 256), dim3(256), 0, stream>>>(
        H, w2T, b2, (void*)Outb, D_, FF_);

    combine_kernel<<<dim3(T_), dim3(256), 0, stream>>>(Outb, row1, row2, wt1, wt2, y);
}
